// SGAT_34729105555801
// MI455X (gfx1250) — matrix operand order repaired and checked by reading, not compiled, not run on hardware
//
#include <hip/hip_runtime.h>
#include <hip/hip_bf16.h>

typedef __attribute__((ext_vector_type(16))) _Float16 v16h;
typedef __attribute__((ext_vector_type(8)))  _Float16 v8h;
typedef __attribute__((ext_vector_type(8)))  float    v8f;
typedef __attribute__((ext_vector_type(4)))  int      i4v;
typedef __attribute__((ext_vector_type(4)))  float    f4v;

#define N_NODES 8192
#define F_IN    256
#define F_OUT   128
#define SLOPE   0.01f
#define NEG_BIG -9.0e15f

// ---------------------------------------------------------------- f32 -> f16
__global__ void cvt_f32_f16(const float* __restrict__ s, _Float16* __restrict__ d, int n) {
    int i = blockIdx.x * blockDim.x + threadIdx.x;
    if (i < n) d[i] = (_Float16)s[i];
}

// ------------------------------------------------- Wh = h @ W  (+ src/dst)
// 512 blocks x 128 threads. Block handles 16 rows, wave w handles cols [32w,32w+32).
__global__ __launch_bounds__(128)
void gat_linear(const _Float16* __restrict__ hf, const _Float16* __restrict__ Wf,
                const float* __restrict__ avec,
                _Float16* __restrict__ Wh16,
                float* __restrict__ srcv, float* __restrict__ dstv) {
    __shared__ float s_src[16], s_dst[16];
    const int tid     = threadIdx.x;
    const int lane    = tid & 31;
    const int wave    = tid >> 5;
    const int rowbase = blockIdx.x * 16;
    const int rowHalf = lane >> 4;      // 0: lanes 0-15, 1: lanes 16-31
    const int mrow    = lane & 15;
    const int b0      = rowHalf ? 8 : 0;

    if (tid < 16) { s_src[tid] = 0.f; s_dst[tid] = 0.f; }
    __syncthreads();

    v8f acc0 = {}; v8f acc1 = {};
    const _Float16* hrow = hf + (size_t)(rowbase + mrow) * F_IN;
    const int n0 = (2 * wave) * 16;
    const int n1 = (2 * wave + 1) * 16;

    for (int kt = 0; kt < F_IN; kt += 32) {
        // A fragment (16x32 f16): lane<16 holds M=lane, K = kt+{0..7,16..23}; lane>=16: K = kt+{8..15,24..31}
        v8h lo = *(const v8h*)(hrow + kt + b0);
        v8h hi = *(const v8h*)(hrow + kt + b0 + 16);
        v16h afrag;
        #pragma unroll
        for (int i = 0; i < 8; ++i) { afrag[i] = lo[i]; afrag[i + 8] = hi[i]; }
        // B fragments (32x16 f16): lane holds column N = lane%16, K in the same order as A
        v16h bf0, bf1;
        #pragma unroll
        for (int i = 0; i < 16; ++i) {
            const int kk = kt + b0 + (i < 8 ? i : i + 8);
            bf0[i] = Wf[(size_t)kk * F_OUT + n0 + mrow];
            bf1[i] = Wf[(size_t)kk * F_OUT + n1 + mrow];
        }
        acc0 = __builtin_amdgcn_wmma_f32_16x16x32_f16(false, afrag, false, bf0, (short)0, acc0, false, false);
        acc1 = __builtin_amdgcn_wmma_f32_16x16x32_f16(false, afrag, false, bf1, (short)0, acc1, false, false);
    }

    // C/D layout: lane 0-15 -> N=lane (M = v), lane 16-31 -> N=lane-16 (M = v+8)
    #pragma unroll
    for (int v = 0; v < 8; ++v) {
        const int r  = rowHalf * 8 + v;
        const int row = rowbase + r;
        const int c0 = n0 + mrow, c1 = n1 + mrow;
        float x0 = acc0[v], x1 = acc1[v];
        Wh16[(size_t)row * F_OUT + c0] = (_Float16)x0;
        Wh16[(size_t)row * F_OUT + c1] = (_Float16)x1;
        atomicAdd(&s_src[r], x0 * avec[c0] + x1 * avec[c1]);
        atomicAdd(&s_dst[r], x0 * avec[F_OUT + c0] + x1 * avec[F_OUT + c1]);
    }
    __syncthreads();
    if (tid < 16) {
        srcv[rowbase + tid] = s_src[tid];
        dstv[rowbase + tid] = s_dst[tid];
    }
}

// --------------------------- fused masked-softmax attention: out = softmax(att) @ Wh
// 512 blocks x 128 threads (4 waves). Each wave: one 16-row tile x 2048-col strip,
// online softmax + WMMA P@V; waves combined flash-decoding style in LDS.
// adj (256 MB, streamed once) is loaded non-temporally so the hot Wh16 working
// set (2 MB, re-read ~1 GB worth across blocks) stays resident in the 192 MB L2.
__global__ __launch_bounds__(128)
void gat_attn(const int* __restrict__ adj, const _Float16* __restrict__ Wh16,
              const float* __restrict__ srcv, const float* __restrict__ dstv,
              float* __restrict__ out) {
    __shared__ float s_acc[16 * F_OUT];            // 8 KB
    __shared__ float s_m[4][16], s_l[4][16], s_f[4][16], s_lg[16];

    const int tid     = threadIdx.x;
    const int lane    = tid & 31;
    const int wave    = tid >> 5;
    const int rowbase = blockIdx.x * 16;
    const int rowHalf = lane >> 4;
    const int mrow    = lane & 15;
    const int b0      = rowHalf ? 8 : 0;

    for (int i = tid; i < 16 * F_OUT; i += 128) s_acc[i] = 0.f;
    __syncthreads();

    const float srow = srcv[rowbase + mrow];
    const int* adjrow = adj + (size_t)(rowbase + mrow) * N_NODES;

    float m_run = -3.0e38f;
    float l_run = 0.f;
    v8f acc[8];
    #pragma unroll
    for (int t = 0; t < 8; ++t) acc[t] = (v8f){};

    const int jbeg = wave * (N_NODES / 4);
    const int jend = jbeg + (N_NODES / 4);

    for (int jt = jbeg; jt < jend; jt += 32) {
        if (jt + 32 < jend) {                      // -> global_prefetch_b8
            __builtin_prefetch(adjrow + jt + 32 + b0, 0, 0);
            __builtin_prefetch(adjrow + jt + 48 + b0, 0, 0);
        }
        // adj (non-temporal: pure stream) + dst for this lane's 16 K values
        i4v a0 = __builtin_nontemporal_load((const i4v*)(adjrow + jt + b0));
        i4v a1 = __builtin_nontemporal_load((const i4v*)(adjrow + jt + b0 + 4));
        i4v a2 = __builtin_nontemporal_load((const i4v*)(adjrow + jt + b0 + 16));
        i4v a3 = __builtin_nontemporal_load((const i4v*)(adjrow + jt + b0 + 20));
        f4v d0 = *(const f4v*)(dstv + jt + b0);
        f4v d1 = *(const f4v*)(dstv + jt + b0 + 4);
        f4v d2 = *(const f4v*)(dstv + jt + b0 + 16);
        f4v d3 = *(const f4v*)(dstv + jt + b0 + 20);

        float e[16];
        {
            auto ec = [&](int ad, float dv) {
                float x = srow + dv;
                float lr = x > 0.f ? x : SLOPE * x;
                return ad > 0 ? lr : NEG_BIG;
            };
            #pragma unroll
            for (int i = 0; i < 4; ++i) {
                e[i]      = ec(a0[i], d0[i]);
                e[i + 4]  = ec(a1[i], d1[i]);
                e[i + 8]  = ec(a2[i], d2[i]);
                e[i + 12] = ec(a3[i], d3[i]);
            }
        }

        float mloc = e[0];
        #pragma unroll
        for (int i = 1; i < 16; ++i) mloc = fmaxf(mloc, e[i]);
        mloc = fmaxf(mloc, __shfl_xor(mloc, 16, 32));   // partner lane holds other half of K
        const float m_new = fmaxf(m_run, mloc);
        const float scale = __expf(m_run - m_new);

        float psum = 0.f;
        v16h afrag;
        #pragma unroll
        for (int i = 0; i < 16; ++i) {
            float p = __expf(e[i] - m_new);
            psum += p;
            afrag[i] = (_Float16)p;
        }
        psum += __shfl_xor(psum, 16, 32);
        l_run = l_run * scale + psum;
        m_run = m_new;

        if (__any(scale < 1.0f)) {                  // wave-uniform; rare after warmup
            #pragma unroll
            for (int v = 0; v < 8; ++v) {
                float sv = __shfl(scale, rowHalf * 8 + v, 32); // scale of row this lane accumulates
                #pragma unroll
                for (int t = 0; t < 8; ++t) acc[t][v] *= sv;
            }
        }

        // P(16x32 f16) @ Wh16(32x128 f16), 8 WMMAs; B lane = column N=lane%16, K in A's order (L2-resident)
        const _Float16* bbase = Wh16 + mrow;
        #pragma unroll
        for (int t = 0; t < 8; ++t) {
            v16h bfrag;
            #pragma unroll
            for (int i = 0; i < 16; ++i)
                bfrag[i] = bbase[(size_t)(jt + b0 + (i < 8 ? i : i + 8)) * F_OUT + t * 16];
            acc[t] = __builtin_amdgcn_wmma_f32_16x16x32_f16(false, afrag, false, bfrag,
                                                            (short)0, acc[t], false, false);
        }
    }

    // ---- combine 4 waves (flash-decoding style)
    if (lane < 16) { s_m[wave][lane] = m_run; s_l[wave][lane] = l_run; }
    __syncthreads();
    if (tid < 16) {
        float mg = s_m[0][tid];
        #pragma unroll
        for (int w = 1; w < 4; ++w) mg = fmaxf(mg, s_m[w][tid]);
        float lg = 0.f;
        #pragma unroll
        for (int w = 0; w < 4; ++w) {
            float f = __expf(s_m[w][tid] - mg);
            s_f[w][tid] = f;
            lg += s_l[w][tid] * f;
        }
        s_lg[tid] = lg;
    }
    __syncthreads();
    #pragma unroll
    for (int t = 0; t < 8; ++t) {
        #pragma unroll
        for (int v = 0; v < 8; ++v) {
            const int r = rowHalf * 8 + v;
            const int c = t * 16 + mrow;
            atomicAdd(&s_acc[r * F_OUT + c], acc[t][v] * s_f[wave][r]);
        }
    }
    __syncthreads();
    for (int i = tid; i < 16 * F_OUT; i += 128) {
        const int r = i >> 7;
        // out is written once, never re-read: non-temporal store
        __builtin_nontemporal_store(s_acc[i] / s_lg[r],
                                    out + (size_t)(rowbase + r) * F_OUT + (i & 127));
    }
}

// ---------------------------------------------------------------- launcher
extern "C" void kernel_launch(void* const* d_in, const int* in_sizes, int n_in,
                              void* d_out, int out_size, void* d_ws, size_t ws_size,
                              hipStream_t stream) {
    const float* h   = (const float*)d_in[0];   // [8192,256]
    const int*   adj = (const int*)  d_in[1];   // [8192,8192]
    const float* W   = (const float*)d_in[2];   // [256,128]
    const float* a   = (const float*)d_in[3];   // [256]
    float* out = (float*)d_out;                 // [8192,128]

    char* ws = (char*)d_ws;
    _Float16* hf16 = (_Float16*)(ws);                                  // 4 MiB
    _Float16* Wf16 = (_Float16*)(ws + (4u << 20));                     // 64 KiB
    _Float16* Wh16 = (_Float16*)(ws + (4u << 20) + (64u << 10));       // 2 MiB
    float*    srcv = (float*)   (ws + (6u << 20) + (64u << 10));       // 32 KiB
    float*    dstv = (float*)   (ws + (6u << 20) + (96u << 10));       // 32 KiB

    cvt_f32_f16<<<(N_NODES * F_IN) / 256, 256, 0, stream>>>(h, hf16, N_NODES * F_IN);
    cvt_f32_f16<<<(F_IN * F_OUT + 255) / 256, 256, 0, stream>>>(W, Wf16, F_IN * F_OUT);
    gat_linear<<<N_NODES / 16, 128, 0, stream>>>(hf16, Wf16, a, Wh16, srcv, dstv);
    gat_attn  <<<N_NODES / 16, 128, 0, stream>>>(adj, Wh16, srcv, dstv, out);
}
